// GPTEncoder_46780783788037
// MI455X (gfx1250) — matrix operand order repaired and checked by reading, not compiled, not run on hardware
//
#include <hip/hip_runtime.h>
#include <math.h>

typedef __attribute__((ext_vector_type(16))) __bf16 v16bf;
typedef __attribute__((ext_vector_type(8)))  float  v8f;
typedef __attribute__((ext_vector_type(8)))  unsigned short v8us;

#define TT    81
#define TPAD  96
#define CC    256
#define LDH   264   // padded bf16 row stride for [96][256] buffers (528B -> conflict-free)
#define LDP   104   // padded stride for P [96][96] and vT [256][96] (208B -> conflict-free)
#define NRECUR 8

// ---------- helpers ----------

__device__ __forceinline__ unsigned short f2bf(float f) {
  unsigned int u = __float_as_uint(f);
  u += 0x7FFFu + ((u >> 16) & 1u);          // round-to-nearest-even
  return (unsigned short)(u >> 16);
}

__device__ __forceinline__ v8us ld8(const unsigned short* p) {
  return *(const v8us*)p;                    // 16B load (ds_load_b128 / global_load_b128)
}

__device__ __forceinline__ v16bf mk16(v8us lo, v8us hi) {
  union { v8us u[2]; v16bf v; } x;
  x.u[0] = lo; x.u[1] = hi;
  return x.v;
}

// A fragment 16x32 bf16 (CDNA5 16-bit A layout).
__device__ __forceinline__ v16bf frag_a(const unsigned short* A, int lda, int m0, int k0, int lane) {
  int r  = m0 + (lane & 15);
  int kh = (lane >> 4) << 3;
  const unsigned short* p = A + r * lda + k0 + kh;
  return mk16(ld8(p), ld8(p + 16));
}

// B fragment 32x16 bf16 from a [N][K]-stored matrix.
__device__ __forceinline__ v16bf frag_b(const unsigned short* Bt, int ldb, int n0, int k0, int lane) {
  int n  = n0 + (lane & 15);
  int ks = k0 + ((lane >> 4) << 3);
  const unsigned short* p = Bt + n * ldb + ks;
  return mk16(ld8(p), ld8(p + 16));
}

__device__ __forceinline__ bool sud_same(int i, int j) {
  int ri = i / 9, ci = i - ri * 9;
  int rj = j / 9, cj = j - rj * 9;
  return (ri == rj) || (ci == cj) || ((ri / 3 == rj / 3) && (ci / 3 == cj / 3));
}

// LayerNorm over C=256 per row -> bf16 out (stride LDH). Wave per row, 8 cols/lane.
__device__ void ln_bf16(const float* __restrict__ xf, const float* __restrict__ g,
                        const float* __restrict__ bsh, unsigned short* __restrict__ out,
                        int lane, int wave) {
  for (int r = wave; r < TT; r += 8) {
    float v[8]; float s = 0.f;
    #pragma unroll
    for (int u = 0; u < 8; ++u) { v[u] = xf[r * CC + lane + 32 * u]; s += v[u]; }
    #pragma unroll
    for (int o = 16; o > 0; o >>= 1) s += __shfl_xor(s, o, 32);
    float mu = s * (1.0f / 256.0f);
    float s2 = 0.f;
    #pragma unroll
    for (int u = 0; u < 8; ++u) { float d = v[u] - mu; s2 += d * d; }
    #pragma unroll
    for (int o = 16; o > 0; o >>= 1) s2 += __shfl_xor(s2, o, 32);
    float inv = rsqrtf(s2 * (1.0f / 256.0f) + 1e-5f);
    #pragma unroll
    for (int u = 0; u < 8; ++u) {
      int c = lane + 32 * u;
      out[r * LDH + c] = f2bf((v[u] - mu) * inv * g[c] + bsh[c]);
    }
  }
}

// [96,256] @ [256,256] GEMM, column-major per wave (columns {wave, wave+8}).
// Per column: K split in two halves; 4 B fragments register-resident per half
// (B still read from L2 exactly once per column), 6 per-M accumulators live
// across halves, and a 2-deep A-fragment double buffer so ds_load latency
// overlaps the WMMA pipe (loads for step i+2 issued right after WMMA i).
// MODE 0: bf16 store (+bias)   MODE 1: bf16 transposed store (+bias)  (for vT)
// MODE 2: residual add into xf (+bias)   MODE 3: exact-GELU -> bf16 store (+bias)
template <int MODE>
__device__ void gemm6x16(const unsigned short* __restrict__ A, int lda,
                         const unsigned short* __restrict__ Bt, int ldb,
                         const float* __restrict__ bias,
                         unsigned short* out, int ostride,
                         float* xf, int lane, int wave) {
  #pragma unroll
  for (int cidx = 0; cidx < 2; ++cidx) {
    const int n0 = (wave + cidx * 8) * 16;
    v8f accs[6] = {};
    #pragma unroll
    for (int khalf = 0; khalf < 2; ++khalf) {
      const int kb = khalf * 128;
      v16bf bfrag[4];
      #pragma unroll
      for (int ks = 0; ks < 4; ++ks) bfrag[ks] = frag_b(Bt, ldb, n0, kb + ks * 32, lane);
      v16bf ap[2];
      ap[0] = frag_a(A, lda, 0, kb, lane);
      ap[1] = frag_a(A, lda, 0, kb + 32, lane);
      #pragma unroll
      for (int i = 0; i < 24; ++i) {              // i = mt*4 + ks
        const int mt = i >> 2;
        accs[mt] = __builtin_amdgcn_wmma_f32_16x16x32_bf16(false, ap[i & 1], false,
                       bfrag[i & 3], (short)0, accs[mt], false, false);
        const int j = i + 2;
        if (j < 24)
          ap[i & 1] = frag_a(A, lda, (j >> 2) * 16, kb + (j & 3) * 32, lane);
      }
    }
    const int n  = n0 + (lane & 15);
    const float bn = bias[n];
    #pragma unroll
    for (int mt = 0; mt < 6; ++mt) {
      const v8f acc = accs[mt];
      const int mb = mt * 16 + ((lane >> 4) << 3);
      if (MODE == 0) {
        #pragma unroll
        for (int r = 0; r < 8; ++r) out[(mb + r) * ostride + n] = f2bf(acc[r] + bn);
      } else if (MODE == 1) {
        #pragma unroll
        for (int r = 0; r < 8; ++r) out[n * ostride + (mb + r)] = f2bf(acc[r] + bn);
      } else if (MODE == 2) {
        #pragma unroll
        for (int r = 0; r < 8; ++r) xf[(mb + r) * CC + n] += acc[r] + bn;
      } else {
        #pragma unroll
        for (int r = 0; r < 8; ++r) {
          float z = acc[r] + bn;
          out[(mb + r) * ostride + n] = f2bf(0.5f * z * (1.0f + erff(z * 0.70710678118f)));
        }
      }
    }
  }
}

// ---------- weight prep: f32 [K][N] -> bf16 [N][K] ----------

__global__ void transpose_bf16_kernel(const float* __restrict__ W,
                                      unsigned short* __restrict__ out, int K, int N) {
  int i = blockIdx.x * blockDim.x + threadIdx.x;
  if (i < K * N) {
    int n = i / K, k = i - n * K;
    out[i] = f2bf(W[k * N + n]);
  }
}

// ---------- main fused kernel: one workgroup per batch element ----------

__global__ __launch_bounds__(256)
void sudoku_gpt_kernel(const int* __restrict__ idx,
                       const float* __restrict__ tok_emb,
                       const float* __restrict__ pos_emb,
                       const float* __restrict__ ln1_g, const float* __restrict__ ln1_b,
                       const float* __restrict__ bq, const float* __restrict__ bk,
                       const float* __restrict__ bv, const float* __restrict__ bp,
                       const float* __restrict__ ln2_g, const float* __restrict__ ln2_b,
                       const float* __restrict__ b1, const float* __restrict__ b2,
                       const float* __restrict__ lnf_g, const float* __restrict__ lnf_b,
                       const float* __restrict__ head_W,
                       const unsigned short* __restrict__ WqT,
                       const unsigned short* __restrict__ WkT,
                       const unsigned short* __restrict__ WvT,
                       const unsigned short* __restrict__ WpT,
                       const unsigned short* __restrict__ W1T,
                       const unsigned short* __restrict__ W2T,
                       float* __restrict__ outHidden, float* __restrict__ outLogits) {
  extern __shared__ char smem[];
  float*          xf   = (float*)(smem);                       // 96*256 f32     (98304 B)
  unsigned short* bufA = (unsigned short*)(smem + 98304);      // h / h2, LDH    (50688 B)
  float*          Sbuf = (float*)(smem + 98304);               // S f32 96x96 overlays dead h
  unsigned short* bufB = (unsigned short*)(smem + 148992);     // q, then y, LDH (50688 B)
  unsigned short* bufC = (unsigned short*)(smem + 199680);     // k, LDH         (50688 B)
  unsigned short* bufD = (unsigned short*)(smem + 250368);     // vT(LDP)/u(LDH) (53248 B)
  unsigned short* Pbuf = (unsigned short*)(smem + 303616);     // P bf16, LDP    (19968 B)
  // total 323584 B = 158 x 2KB blocks <= 320 KB/WGP

  const int b    = blockIdx.x;
  const int tid  = threadIdx.x;
  const int lane = tid & 31;
  const int wave = tid >> 5;

  // embedding: x = tok_emb[idx] + pos_emb
  for (int i = tid; i < TT * CC; i += 256) {
    int t = i >> 8, c = i & 255;
    xf[i] = tok_emb[idx[b * TT + t] * CC + c] + pos_emb[i];
  }
  __syncthreads();

  for (int it = 0; it < NRECUR; ++it) {
    // h = LN1(x)
    ln_bf16(xf, ln1_g, ln1_b, bufA, lane, wave);
    __syncthreads();

    // q,k,v projections (v stored transposed [C][Tpad] with stride LDP)
    gemm6x16<0>(bufA, LDH, WqT, CC, bq, bufB, LDH, nullptr, lane, wave);
    gemm6x16<0>(bufA, LDH, WkT, CC, bk, bufC, LDH, nullptr, lane, wave);
    gemm6x16<1>(bufA, LDH, WvT, CC, bv, bufD, LDP, nullptr, lane, wave);
    __syncthreads();

    // attention, one head at a time
    for (int hh = 0; hh < 4; ++hh) {
      // S = q @ k^T  (k stored [t][C] is exactly the [N][K] form for the B operand)
      for (int t = wave; t < 36; t += 8) {
        int m0 = (t % 6) * 16, n0 = (t / 6) * 16;
        v16bf a0 = frag_a(bufB, LDH, m0, hh * 64, lane);
        v16bf b0 = frag_b(bufC, LDH, n0, hh * 64, lane);
        v16bf a1 = frag_a(bufB, LDH, m0, hh * 64 + 32, lane);
        v16bf b1 = frag_b(bufC, LDH, n0, hh * 64 + 32, lane);
        v8f acc = {};
        acc = __builtin_amdgcn_wmma_f32_16x16x32_bf16(false, a0, false, b0, (short)0, acc,
                                                      false, false);
        acc = __builtin_amdgcn_wmma_f32_16x16x32_bf16(false, a1, false, b1, (short)0, acc,
                                                      false, false);
        int n = n0 + (lane & 15), mb = m0 + ((lane >> 4) << 3);
        #pragma unroll
        for (int r = 0; r < 8; ++r) Sbuf[(mb + r) * TPAD + n] = acc[r];
      }
      __syncthreads();

      // masked softmax (sudoku mask computed analytically), scale = 1/sqrt(64)
      for (int r = wave; r < TT; r += 8) {
        float sv[3];
        #pragma unroll
        for (int u = 0; u < 3; ++u) {
          int j = lane + 32 * u;
          sv[u] = (j < TT && sud_same(r, j)) ? Sbuf[r * TPAD + j] * 0.125f : -3.0e38f;
        }
        float mx = fmaxf(fmaxf(sv[0], sv[1]), sv[2]);
        #pragma unroll
        for (int o = 16; o > 0; o >>= 1) mx = fmaxf(mx, __shfl_xor(mx, o, 32));
        float es[3]; float sum = 0.f;
        #pragma unroll
        for (int u = 0; u < 3; ++u) { es[u] = expf(sv[u] - mx); sum += es[u]; }
        #pragma unroll
        for (int o = 16; o > 0; o >>= 1) sum += __shfl_xor(sum, o, 32);
        float rinv = 1.0f / sum;
        #pragma unroll
        for (int u = 0; u < 3; ++u) Pbuf[r * LDP + lane + 32 * u] = f2bf(es[u] * rinv);
      }
      __syncthreads();

      // y_h = P @ V  -> overwrite dead q columns of this head in bufB
      for (int t = wave; t < 24; t += 8) {
        int m0 = (t % 6) * 16, n0 = (t / 6) * 16;
        v16bf af[3], bf[3];
        #pragma unroll
        for (int ks = 0; ks < 3; ++ks) {
          af[ks] = frag_a(Pbuf, LDP, m0, ks * 32, lane);
          bf[ks] = frag_b(bufD, LDP, hh * 64 + n0, ks * 32, lane);
        }
        v8f acc = {};
        #pragma unroll
        for (int ks = 0; ks < 3; ++ks)
          acc = __builtin_amdgcn_wmma_f32_16x16x32_bf16(false, af[ks], false, bf[ks],
                                                        (short)0, acc, false, false);
        int n = hh * 64 + n0 + (lane & 15), mb = m0 + ((lane >> 4) << 3);
        #pragma unroll
        for (int r = 0; r < 8; ++r) bufB[(mb + r) * LDH + n] = f2bf(acc[r]);
      }
      __syncthreads();
    }

    // x += y @ Wp + bp
    gemm6x16<2>(bufB, LDH, WpT, CC, bp, nullptr, 0, xf, lane, wave);
    __syncthreads();

    // h2 = LN2(x)
    ln_bf16(xf, ln2_g, ln2_b, bufA, lane, wave);
    __syncthreads();

    // MLP in 4 chunks of 256 hidden units; persistent f32 accumulators for MLP2,
    // half-K register-resident B fragments + 2-deep A double buffer.
    v8f acc2[12] = {};            // [cidx*6 + mt]

    for (int ch = 0; ch < 4; ++ch) {
      // u = gelu(h2 @ W1[:, chunk] + b1[chunk])  -> bufD bf16 (stride LDH)
      gemm6x16<3>(bufA, LDH, W1T + ch * 256 * CC, CC, b1 + ch * 256,
                  bufD, LDH, nullptr, lane, wave);
      __syncthreads();
      // acc2 += u @ W2[chunk, :]
      #pragma unroll
      for (int cidx = 0; cidx < 2; ++cidx) {
        const int n0 = (wave + cidx * 8) * 16;
        #pragma unroll
        for (int khalf = 0; khalf < 2; ++khalf) {
          const int ka = khalf * 128;            // A-side K offset (within chunk)
          const int kbb = ch * 256 + ka;         // B-side K offset (global)
          v16bf bfrag[4];
          #pragma unroll
          for (int ks = 0; ks < 4; ++ks)
            bfrag[ks] = frag_b(W2T, 1024, n0, kbb + ks * 32, lane);
          v16bf ap[2];
          ap[0] = frag_a(bufD, LDH, 0, ka, lane);
          ap[1] = frag_a(bufD, LDH, 0, ka + 32, lane);
          #pragma unroll
          for (int i = 0; i < 24; ++i) {         // i = mt*4 + ks
            const int mt = i >> 2;
            acc2[cidx * 6 + mt] = __builtin_amdgcn_wmma_f32_16x16x32_bf16(
                false, ap[i & 1], false, bfrag[i & 3], (short)0, acc2[cidx * 6 + mt],
                false, false);
            const int j = i + 2;
            if (j < 24)
              ap[i & 1] = frag_a(bufD, LDH, (j >> 2) * 16, ka + (j & 3) * 32, lane);
          }
        }
      }
      __syncthreads();
    }
    // x += mlp_out + b2
    #pragma unroll
    for (int cidx = 0; cidx < 2; ++cidx) {
      const int n0 = (wave + cidx * 8) * 16;
      const int n  = n0 + (lane & 15);
      const float bn = b2[n];
      #pragma unroll
      for (int mt = 0; mt < 6; ++mt) {
        const int mb = mt * 16 + ((lane >> 4) << 3);
        #pragma unroll
        for (int r = 0; r < 8; ++r) xf[(mb + r) * CC + n] += acc2[cidx * 6 + mt][r] + bn;
      }
    }
    __syncthreads();
  }

  // hidden = LN_f(x): in-place + store to global
  for (int r = wave; r < TT; r += 8) {
    float v[8]; float s = 0.f;
    #pragma unroll
    for (int u = 0; u < 8; ++u) { v[u] = xf[r * CC + lane + 32 * u]; s += v[u]; }
    #pragma unroll
    for (int o = 16; o > 0; o >>= 1) s += __shfl_xor(s, o, 32);
    float mu = s * (1.0f / 256.0f);
    float s2 = 0.f;
    #pragma unroll
    for (int u = 0; u < 8; ++u) { float d = v[u] - mu; s2 += d * d; }
    #pragma unroll
    for (int o = 16; o > 0; o >>= 1) s2 += __shfl_xor(s2, o, 32);
    float inv = rsqrtf(s2 * (1.0f / 256.0f) + 1e-5f);
    #pragma unroll
    for (int u = 0; u < 8; ++u) {
      int c = lane + 32 * u;
      float hv = (v[u] - mu) * inv * lnf_g[c] + lnf_b[c];
      xf[r * CC + c] = hv;
      outHidden[(size_t)b * TT * CC + r * CC + c] = hv;
    }
  }
  __syncthreads();

  // logits = hidden @ head_W, stored permuted as [B, 9, T, 1]
  for (int o = tid; o < TT * 9; o += 256) {
    int t = o / 9, n = o - t * 9;
    float s = 0.f;
    for (int c = 0; c < CC; ++c) s += xf[t * CC + c] * head_W[c * 9 + n];
    outLogits[(size_t)b * (9 * TT) + n * TT + t] = s;
  }
}

// ---------- host launch ----------

extern "C" void kernel_launch(void* const* d_in, const int* in_sizes, int n_in,
                              void* d_out, int out_size, void* d_ws, size_t ws_size,
                              hipStream_t stream) {
  const int*   idx     = (const int*)d_in[0];
  const float* tok_emb = (const float*)d_in[1];
  const float* pos_emb = (const float*)d_in[2];
  const float* ln1_g   = (const float*)d_in[3];
  const float* ln1_b   = (const float*)d_in[4];
  const float* Wq = (const float*)d_in[5];  const float* bq = (const float*)d_in[6];
  const float* Wk = (const float*)d_in[7];  const float* bk = (const float*)d_in[8];
  const float* Wv = (const float*)d_in[9];  const float* bv = (const float*)d_in[10];
  const float* Wp = (const float*)d_in[11]; const float* bp = (const float*)d_in[12];
  const float* ln2_g = (const float*)d_in[13]; const float* ln2_b = (const float*)d_in[14];
  const float* W1 = (const float*)d_in[15]; const float* b1 = (const float*)d_in[16];
  const float* W2 = (const float*)d_in[17]; const float* b2 = (const float*)d_in[18];
  const float* lnf_g = (const float*)d_in[19]; const float* lnf_b = (const float*)d_in[20];
  const float* head_W = (const float*)d_in[21];

  // bf16 transposed weights in workspace
  unsigned short* WqT = (unsigned short*)d_ws;
  unsigned short* WkT = WqT + 256 * 256;
  unsigned short* WvT = WkT + 256 * 256;
  unsigned short* WpT = WvT + 256 * 256;
  unsigned short* W1T = WpT + 256 * 256;      // [1024][256]
  unsigned short* W2T = W1T + 1024 * 256;     // [256][1024]

  transpose_bf16_kernel<<<(256 * 256 + 255) / 256, 256, 0, stream>>>(Wq, WqT, 256, 256);
  transpose_bf16_kernel<<<(256 * 256 + 255) / 256, 256, 0, stream>>>(Wk, WkT, 256, 256);
  transpose_bf16_kernel<<<(256 * 256 + 255) / 256, 256, 0, stream>>>(Wv, WvT, 256, 256);
  transpose_bf16_kernel<<<(256 * 256 + 255) / 256, 256, 0, stream>>>(Wp, WpT, 256, 256);
  transpose_bf16_kernel<<<(256 * 1024 + 255) / 256, 256, 0, stream>>>(W1, W1T, 256, 1024);
  transpose_bf16_kernel<<<(1024 * 256 + 255) / 256, 256, 0, stream>>>(W2, W2T, 1024, 256);

  const int B = 1024;
  float* outHidden = (float*)d_out;
  float* outLogits = outHidden + (size_t)B * TT * CC;
  const size_t shmem = 323584;  // 158 x 2KB LDS blocks, <= 320 KB/WGP

  sudoku_gpt_kernel<<<B, 256, shmem, stream>>>(
      idx, tok_emb, pos_emb, ln1_g, ln1_b, bq, bk, bv, bp, ln2_g, ln2_b,
      b1, b2, lnf_g, lnf_b, head_W, WqT, WkT, WvT, WpT, W1T, W2T,
      outHidden, outLogits);
}
